// CLIPLoss_20753281974299
// MI455X (gfx1250) — matrix operand order repaired and checked by reading, not compiled, not run on hardware
//
#include <hip/hip_runtime.h>
#include <hip/hip_bf16.h>

// ---------------------------------------------------------------------------
// CLIP loss, fused flash-style LSE on CDNA5 (gfx1250, wave32, WMMA bf16).
//   loss = mean_i( 0.5*(lse_row[i] + lse_col[i]) - scale*dot(img_i, txt_i) )
// ---------------------------------------------------------------------------

typedef __attribute__((ext_vector_type(16))) __bf16        v16bf;
typedef __attribute__((ext_vector_type(8)))  float         v8f;
typedef __attribute__((ext_vector_type(4)))  unsigned int  uint4v;

#define D_DIM   1024
#define TILE_M  128
#define TILE_N  128

union FragU { uint4v u[2]; v16bf v; };

// A fragment: 16x32 bf16 tile, row-major source [rows x D].
// lane m = lane&15 owns row m; half = lane>>4.
// regs 0-3 <- K = k0 + 8*half .. +7 ; regs 4-7 <- K = k0 + 16 + 8*half .. +7
__device__ inline v16bf load_a_frag(const unsigned short* __restrict__ base,
                                    int row0, int k0, int lane) {
    int m = lane & 15, h = lane >> 4;
    const unsigned short* p = base + (size_t)(row0 + m) * D_DIM + k0 + 8 * h;
    FragU f;
    f.u[0] = *(const uint4v*)p;        // 8 bf16
    f.u[1] = *(const uint4v*)(p + 16); // 8 bf16
    return f.v;
}

// B fragment: 32x16 bf16 tile (KxN) sourced from row-major txt [rows x D],
// i.e. B(k,n) = T[row0+n, k0+k].  lane n = lane&15; half = lane>>4.
// regs 0-3 <- K = k0 + 8*half .. +7 ; regs 4-7 <- K = k0 + 16 + 8*half .. +7
__device__ inline v16bf load_b_frag(const unsigned short* __restrict__ base,
                                    int row0, int k0, int lane) {
    int n = lane & 15, h = lane >> 4;
    const unsigned short* p = base + (size_t)(row0 + n) * D_DIM + k0 + 8 * h;
    FragU f;
    f.u[0] = *(const uint4v*)p;
    f.u[1] = *(const uint4v*)(p + 16);
    return f.v;
}

// ---------------------------------------------------------------------------
// fp32 -> bf16 (round-to-nearest-even), 4 floats per thread.
// ---------------------------------------------------------------------------
__device__ inline unsigned int f2bf(float f) {
    unsigned int u = __float_as_uint(f);
    u += 0x7FFFu + ((u >> 16) & 1u);
    return u >> 16;
}

__global__ __launch_bounds__(256)
void cvt_bf16_kernel(const float* __restrict__ src,
                     unsigned short* __restrict__ dst, long long n4) {
    long long i = (long long)blockIdx.x * 256 + threadIdx.x;
    if (i >= n4) return;
    float4 v = ((const float4*)src)[i];
    unsigned long long packed =
        (unsigned long long)f2bf(v.x)        |
        ((unsigned long long)f2bf(v.y) << 16) |
        ((unsigned long long)f2bf(v.z) << 32) |
        ((unsigned long long)f2bf(v.w) << 48);
    ((unsigned long long*)dst)[i] = packed;
}

// ---------------------------------------------------------------------------
// diag[i] = scale * dot(img[i], txt[i])  (fp32, one wave per row)
// ---------------------------------------------------------------------------
__global__ __launch_bounds__(256)
void diag_kernel(const float* __restrict__ img, const float* __restrict__ txt,
                 const float* __restrict__ scale_ptr, float* __restrict__ diag) {
    int row  = blockIdx.x * 8 + (threadIdx.x >> 5);
    int lane = threadIdx.x & 31;
    const float* a = img + (size_t)row * D_DIM;
    const float* b = txt + (size_t)row * D_DIM;
    float s = 0.f;
    for (int k = lane * 4; k < D_DIM; k += 32 * 4) {
        float4 x = *(const float4*)(a + k);
        float4 y = *(const float4*)(b + k);
        s += x.x * y.x + x.y * y.y + x.z * y.z + x.w * y.w;
    }
    for (int off = 16; off; off >>= 1) s += __shfl_xor(s, off, 32);
    if (lane == 0) diag[row] = s * (*scale_ptr);
}

// ---------------------------------------------------------------------------
// Fused GEMM + online logsumexp over rows of  scale * A @ B^T.
// Grid: N/128 blocks of 256 threads (8 waves, 4(M) x 2(N) wave grid).
// Each wave: 32x64 region = 2x4 grid of 16x16 WMMA accumulators.
// ---------------------------------------------------------------------------
__global__ __launch_bounds__(256)
void lse_rows_kernel(const unsigned short* __restrict__ A,
                     const unsigned short* __restrict__ B,
                     const float* __restrict__ scale_ptr,
                     float* __restrict__ lse_out, int N) {
    __shared__ float sm_m[TILE_M][2];
    __shared__ float sm_s[TILE_M][2];
    __shared__ float run_m[TILE_M];
    __shared__ float run_s[TILE_M];

    const int tid  = threadIdx.x;
    const int lane = tid & 31;
    const int wave = tid >> 5;
    const int wm   = wave & 3;   // 0..3 along M (32 rows each)
    const int wn   = wave >> 2;  // 0..1 along N (64 cols each)
    const int rowBase = blockIdx.x * TILE_M;
    const float scale = *scale_ptr;

    if (tid < TILE_M) { run_m[tid] = -__builtin_inff(); run_s[tid] = 0.f; }
    __syncthreads();

    for (int j0 = 0; j0 < N; j0 += TILE_N) {
        v8f acc[2][4] = {};

        for (int k0 = 0; k0 < D_DIM; k0 += 32) {
            v16bf a[2], b[4];
#pragma unroll
            for (int mt = 0; mt < 2; ++mt)
                a[mt] = load_a_frag(A, rowBase + wm * 32 + mt * 16, k0, lane);
#pragma unroll
            for (int nt = 0; nt < 4; ++nt)
                b[nt] = load_b_frag(B, j0 + wn * 64 + nt * 16, k0, lane);
#pragma unroll
            for (int mt = 0; mt < 2; ++mt)
#pragma unroll
                for (int nt = 0; nt < 4; ++nt)
                    acc[mt][nt] = __builtin_amdgcn_wmma_f32_16x16x32_bf16(
                        false, a[mt], false, b[nt],
                        (short)0, acc[mt][nt], false, false);
        }

        // Per-row (max, sum exp) over this 128-wide tile.
        // C layout: VGPR g, lanes 0-15: (M=g, N=lane); lanes 16-31: (M=g+8, N=lane-16)
#pragma unroll
        for (int mt = 0; mt < 2; ++mt) {
#pragma unroll
            for (int g = 0; g < 8; ++g) {
                float mx = -__builtin_inff();
#pragma unroll
                for (int nt = 0; nt < 4; ++nt)
                    mx = fmaxf(mx, scale * acc[mt][nt][g]);
                for (int off = 1; off < 16; off <<= 1)
                    mx = fmaxf(mx, __shfl_xor(mx, off, 32));
                float sum = 0.f;
#pragma unroll
                for (int nt = 0; nt < 4; ++nt)
                    sum += __expf(scale * acc[mt][nt][g] - mx);
                for (int off = 1; off < 16; off <<= 1)
                    sum += __shfl_xor(sum, off, 32);
                if ((lane & 15) == 0) {
                    int r = wm * 32 + mt * 16 + g + 8 * (lane >> 4);
                    sm_m[r][wn] = mx;
                    sm_s[r][wn] = sum;
                }
            }
        }
        __syncthreads();
        if (tid < TILE_M) {
            float m0 = run_m[tid], s0 = run_s[tid];
            float ma = sm_m[tid][0], sa = sm_s[tid][0];
            float mb = sm_m[tid][1], sb = sm_s[tid][1];
            float mn = fmaxf(m0, fmaxf(ma, mb));
            float sn = s0 * __expf(m0 - mn) + sa * __expf(ma - mn) + sb * __expf(mb - mn);
            run_m[tid] = mn; run_s[tid] = sn;
        }
        __syncthreads();
    }

    if (tid < TILE_M)
        lse_out[rowBase + tid] = run_m[tid] + __logf(run_s[tid]);
}

// ---------------------------------------------------------------------------
// loss = (1/N) * sum_i ( 0.5*(lse_i[i] + lse_t[i]) - diag[i] )
// ---------------------------------------------------------------------------
__global__ __launch_bounds__(256)
void finalize_kernel(const float* __restrict__ lse_i,
                     const float* __restrict__ lse_t,
                     const float* __restrict__ diag,
                     float* __restrict__ out, int N) {
    __shared__ float sm[8];
    float s = 0.f;
    for (int i = threadIdx.x; i < N; i += 256)
        s += 0.5f * (lse_i[i] + lse_t[i]) - diag[i];
    for (int off = 16; off; off >>= 1) s += __shfl_xor(s, off, 32);
    if ((threadIdx.x & 31) == 0) sm[threadIdx.x >> 5] = s;
    __syncthreads();
    if (threadIdx.x == 0) {
        float v = 0.f;
        for (int i = 0; i < 8; ++i) v += sm[i];
        out[0] = v / (float)N;
    }
}

// ---------------------------------------------------------------------------
extern "C" void kernel_launch(void* const* d_in, const int* in_sizes, int n_in,
                              void* d_out, int out_size, void* d_ws, size_t ws_size,
                              hipStream_t stream) {
    const float* img   = (const float*)d_in[0];
    const float* txt   = (const float*)d_in[1];
    const float* scale = (const float*)d_in[2];
    float* out = (float*)d_out;

    const int N = in_sizes[0] / D_DIM;   // 16384

    // workspace layout
    unsigned short* imgb = (unsigned short*)d_ws;
    unsigned short* txtb = imgb + (size_t)N * D_DIM;
    float* lse_i = (float*)(txtb + (size_t)N * D_DIM);
    float* lse_t = lse_i + N;
    float* diag  = lse_t + N;

    long long n4 = (long long)N * D_DIM / 4;
    int cvtBlocks = (int)((n4 + 255) / 256);
    cvt_bf16_kernel<<<cvtBlocks, 256, 0, stream>>>(img, imgb, n4);
    cvt_bf16_kernel<<<cvtBlocks, 256, 0, stream>>>(txt, txtb, n4);

    diag_kernel<<<N / 8, 256, 0, stream>>>(img, txt, scale, diag);

    // rows of logits (i2t) and rows of logits^T (t2i)
    lse_rows_kernel<<<N / TILE_M, 256, 0, stream>>>(imgb, txtb, scale, lse_i, N);
    lse_rows_kernel<<<N / TILE_M, 256, 0, stream>>>(txtb, imgb, scale, lse_t, N);

    finalize_kernel<<<1, 256, 0, stream>>>(lse_i, lse_t, diag, out, N);
}
